// Block_77721728188533
// MI455X (gfx1250) — matrix operand order repaired and checked by reading, not compiled, not run on hardware
//
#include <hip/hip_runtime.h>
#include <math.h>

// ---------------- model constants ----------------
#define BATCH   4
#define SEQ     2048
#define DM      1024      // d_model
#define DI      2048      // d_inner
#define DS      16        // d_state
#define DTR     64        // dt_rank
#define NTOK    (BATCH * SEQ)   // 8192 tokens
#define XPN     (DTR + 2 * DS)  // 96

// ---------------- types ----------------
typedef __attribute__((ext_vector_type(16))) __bf16 v16bf;
typedef __attribute__((ext_vector_type(8)))  __bf16 v8bf;
typedef __attribute__((ext_vector_type(8)))  float  v8f;
typedef __attribute__((ext_vector_type(4)))  unsigned int u32x4;
typedef __attribute__((ext_vector_type(8)))  int i32x8;
typedef __attribute__((ext_vector_type(4)))  int i32x4;

// ---------------- TDM helper: 2D tile (bf16) global -> LDS ----------------
// Builds D# per CDNA5 ISA ch.8: group0 = {count,lds_addr,global_addr,type=2},
// group1 = {data_size=2B, tensor dims/strides in elements, tile dims}.
__device__ __forceinline__ unsigned lds_off(const void* p) {
  return (unsigned)(unsigned long long)p;   // LDS aperture: low 32 bits = DS offset
}

__device__ __forceinline__ void tdm_load_2d_bf16(
    unsigned lds_byte, const __bf16* gptr,
    int rows_rem, int cols_rem, int row_stride_elems,
    int tile_rows, int tile_cols) {
  unsigned long long ga = (unsigned long long)gptr;
  u32x4 g0;
  g0[0] = 1u;                                              // count=1, user mode
  g0[1] = lds_byte;                                        // lds_addr[31:0]
  g0[2] = (unsigned)(ga & 0xffffffffull);                  // global_addr[31:0]
  g0[3] = (unsigned)((ga >> 32) & 0x01ffffffull)           // global_addr[56:32]
        | (2u << 30);                                      // type=2 (image)
  if (rows_rem < 0) rows_rem = 0;
  if (cols_rem < 0) cols_rem = 0;
  i32x8 g1;
  g1[0] = 0x10000;                                         // data_size=1 -> 2 bytes
  g1[1] = (int)(((unsigned)cols_rem & 0xffffu) << 16);     // tensor_dim0[15:0] @ 63:48
  g1[2] = (int)((((unsigned)cols_rem >> 16) & 0xffffu)     // tensor_dim0[31:16]
        |       (((unsigned)rows_rem & 0xffffu) << 16));   // tensor_dim1[15:0]
  g1[3] = (int)((((unsigned)rows_rem >> 16) & 0xffffu)     // tensor_dim1[31:16]
        |       (((unsigned)tile_cols & 0xffffu) << 16));  // tile_dim0 @127:112
  g1[4] = (int)((unsigned)tile_rows & 0xffffu);            // tile_dim1; tile_dim2=0
  g1[5] = row_stride_elems;                                // tensor_dim0_stride[31:0]
  g1[6] = 0;                                               // stride[47:32], dim1_stride lo
  g1[7] = 0;
  i32x4 z4 = {0, 0, 0, 0};
  i32x8 z8 = {0, 0, 0, 0, 0, 0, 0, 0};
  __builtin_amdgcn_tensor_load_to_lds(g0, g1, z4, z4, z8, 0);
}

// ---------------- GEMM: C[M,N] = A[M,K](lda,bf16) @ W[N,K]^T(bf16) ----------------
// macro-tile 128x128x32, 8 waves, wave tile 32x64, TDM double-buffered staging.
#define BM 128
#define BN 128
#define BK 32

enum { EP_STORE = 0, EP_ACCUM = 1, EP_SOFTPLUS = 2 };

template <int EPI, bool MIRROR>
__global__ __launch_bounds__(256)
void gemm_wmma_bf16(const __bf16* __restrict__ A, const __bf16* __restrict__ W,
                    float* __restrict__ C, __bf16* __restrict__ Cb,
                    const float* __restrict__ bias,
                    int M, int N, int K, int lda) {
  __shared__ __align__(16) __bf16 As[2][BM * BK];
  __shared__ __align__(16) __bf16 Bs[2][BN * BK];

  const int tid  = threadIdx.x;
  const int lane = tid & 31;
  const int wave = tid >> 5;
  const int wm   = wave & 3;   // 4 wave-rows of 32
  const int wn   = wave >> 2;  // 2 wave-cols of 64
  const int m0   = blockIdx.y * BM;
  const int n0   = blockIdx.x * BN;
  const bool issuer = (wave == 0);

  v8f acc[2][4] = {};
  const int nk = K / BK;

  auto issue = [&](int stage, int k0) {
    if (issuer) {
      tdm_load_2d_bf16(lds_off(&As[stage][0]), A + (size_t)m0 * lda + k0,
                       M - m0, K - k0, lda, BM, BK);
      tdm_load_2d_bf16(lds_off(&Bs[stage][0]), W + (size_t)n0 * K + k0,
                       N - n0, K - k0, K, BN, BK);
    }
  };

  issue(0, 0);
  for (int i = 0; i < nk; ++i) {
    const int cur = i & 1;
    if (i + 1 < nk) {
      issue(1 - cur, (i + 1) * BK);
      if (issuer) __builtin_amdgcn_s_wait_tensorcnt((short)2);  // stage i landed
    } else {
      if (issuer) __builtin_amdgcn_s_wait_tensorcnt((short)0);
    }
    __syncthreads();

    // A fragments: lane<16 -> K0..7 & K16..23 ; lane>=16 -> K8..15 & K24..31
    const int ak = (lane < 16) ? 0 : 8;
    v16bf afrag[2];
#pragma unroll
    for (int tm = 0; tm < 2; ++tm) {
      int arow = wm * 32 + tm * 16 + (lane & 15);
      v8bf alo = *(const v8bf*)&As[cur][arow * BK + ak];
      v8bf ahi = *(const v8bf*)&As[cur][arow * BK + 16 + ak];
#pragma unroll
      for (int e = 0; e < 8; ++e) { afrag[tm][e] = alo[e]; afrag[tm][e + 8] = ahi[e]; }
    }

    const int bk = (lane < 16) ? 0 : 8;
#pragma unroll
    for (int tn = 0; tn < 4; ++tn) {
      int bcol = wn * 64 + tn * 16 + (lane & 15);
      v8bf blo = *(const v8bf*)&Bs[cur][bcol * BK + bk];
      v8bf bhi = *(const v8bf*)&Bs[cur][bcol * BK + bk + 16];
      v16bf bfrag;
#pragma unroll
      for (int e = 0; e < 8; ++e) { bfrag[e] = blo[e]; bfrag[e + 8] = bhi[e]; }
#pragma unroll
      for (int tm = 0; tm < 2; ++tm) {
        acc[tm][tn] = __builtin_amdgcn_wmma_f32_16x16x32_bf16(
            false, afrag[tm], false, bfrag, (short)0, acc[tm][tn], false, false);
      }
    }
    __syncthreads();
  }

  // epilogue: C/D layout -> lane<16: rows j; lanes>=16: rows j+8; col = lane&15
#pragma unroll
  for (int tm = 0; tm < 2; ++tm) {
    const int mrow0 = m0 + wm * 32 + tm * 16 + ((lane < 16) ? 0 : 8);
#pragma unroll
    for (int tn = 0; tn < 4; ++tn) {
      int n = n0 + wn * 64 + tn * 16 + (lane & 15);
      if (n < N) {
#pragma unroll
        for (int j = 0; j < 8; ++j) {
          size_t off = (size_t)(mrow0 + j) * N + n;
          float v = acc[tm][tn][j];
          if (EPI == EP_SOFTPLUS) {
            v += bias[n];
            v = (v > 20.0f) ? v : log1pf(expf(v));
          }
          if (EPI == EP_ACCUM) C[off] += v;
          else                 C[off] = v;
          if (MIRROR) Cb[off] = (__bf16)v;
        }
      }
    }
  }
}

// ---------------- fp32 -> bf16 elementwise ----------------
__global__ __launch_bounds__(256)
void cvt_bf16_kernel(const float* __restrict__ src, __bf16* __restrict__ dst, size_t n) {
  size_t i = ((size_t)blockIdx.x * 256 + threadIdx.x) * 4;
  if (i + 3 < n) {
#pragma unroll
    for (int j = 0; j < 4; ++j) dst[i + j] = (__bf16)src[i + j];
  } else {
    for (; i < n; ++i) dst[i] = (__bf16)src[i];
  }
}

// ---------------- LayerNorm (1024), one token per block, bf16 out ----------------
__global__ __launch_bounds__(256)
void layernorm_kernel(const float* __restrict__ x, const float* __restrict__ g,
                      const float* __restrict__ b, __bf16* __restrict__ u) {
  __shared__ float red[256];
  const size_t tok = blockIdx.x;
  const float* xr = x + tok * DM;
  float s = 0.f;
  for (int i = threadIdx.x; i < DM; i += 256) s += xr[i];
  red[threadIdx.x] = s; __syncthreads();
  for (int o = 128; o > 0; o >>= 1) {
    if (threadIdx.x < o) red[threadIdx.x] += red[threadIdx.x + o];
    __syncthreads();
  }
  float mean = red[0] * (1.0f / DM);
  __syncthreads();
  float v = 0.f;
  for (int i = threadIdx.x; i < DM; i += 256) { float d = xr[i] - mean; v += d * d; }
  red[threadIdx.x] = v; __syncthreads();
  for (int o = 128; o > 0; o >>= 1) {
    if (threadIdx.x < o) red[threadIdx.x] += red[threadIdx.x + o];
    __syncthreads();
  }
  float rstd = rsqrtf(red[0] * (1.0f / DM) + 1e-5f);
  for (int i = threadIdx.x; i < DM; i += 256)
    u[tok * DM + i] = (__bf16)((xr[i] - mean) * rstd * g[i] + b[i]);
}

// ---------------- depthwise causal/anti-causal conv + SiLU (f32 + bf16 out) ----------------
__global__ __launch_bounds__(256)
void conv_silu_kernel(const float* __restrict__ xz, const float* __restrict__ cw,
                      const float* __restrict__ cb, float* __restrict__ xcf,
                      __bf16* __restrict__ xcb, int rev) {
  size_t idx = (size_t)blockIdx.x * 256 + threadIdx.x;  // over B*L*DI
  int d = (int)(idx & (DI - 1));
  size_t bl = idx >> 11;
  int l = (int)(bl & (SEQ - 1));
  int b = (int)(bl >> 11);
  float acc = cb[d];
#pragma unroll
  for (int k = 0; k < 4; ++k) {
    int lm = rev ? (l + 3 - k) : (l - 3 + k);
    if (lm >= 0 && lm < SEQ)
      acc += cw[d * 4 + k] * xz[((size_t)b * SEQ + lm) * (2 * DI) + d];
  }
  float s = acc / (1.0f + __expf(-acc));
  xcf[idx] = s;
  xcb[idx] = (__bf16)s;
}

// ---------------- selective scan: one lane per (b,d), bf16 y out ----------------
__global__ __launch_bounds__(256)
void scan_kernel(const float* __restrict__ dt, const float* __restrict__ dbl,
                 const float* __restrict__ xc, const float* __restrict__ xz,
                 const float* __restrict__ A_log, const float* __restrict__ Dp,
                 __bf16* __restrict__ y, int rev) {
  const int b = blockIdx.x >> 3;                        // uniform
  const int d = ((blockIdx.x & 7) << 8) + threadIdx.x;  // 0..2047
  float Areg[DS], h[DS];
#pragma unroll
  for (int n = 0; n < DS; ++n) {
    Areg[n] = -__expf(A_log[d * DS + n]);
    h[n] = 0.f;
  }
  const float Dv = Dp[d];
  for (int t = 0; t < SEQ; ++t) {
    const int tm = rev ? (SEQ - 1 - t) : t;             // uniform
    const size_t tok = (size_t)b * SEQ + tm;            // uniform -> B/C scalarize
    const float dtv = dt[tok * DI + d];
    const float xv  = xc[tok * DI + d];
    const float zv  = xz[tok * (2 * DI) + DI + d];
    const float* bc = dbl + tok * XPN;
    const float dx = dtv * xv;
    float acc = 0.f;
#pragma unroll
    for (int n = 0; n < DS; ++n) {
      float dA = __expf(dtv * Areg[n]);
      h[n] = h[n] * dA + dx * bc[DTR + n];
      acc += h[n] * bc[DTR + DS + n];
    }
    float yo = acc + Dv * xv;
    float sz = zv / (1.0f + __expf(-zv));
    y[tok * DI + d] = (__bf16)(yo * sz);
  }
}

// ---------------- out = x ----------------
__global__ __launch_bounds__(256)
void copy_kernel(const float* __restrict__ src, float* __restrict__ dst, size_t n) {
  size_t i = (size_t)blockIdx.x * 256 + threadIdx.x;
  if (i < n) dst[i] = src[i];
}

// ---------------- host launcher ----------------
extern "C" void kernel_launch(void* const* d_in, const int* in_sizes, int n_in,
                              void* d_out, int out_size, void* d_ws, size_t ws_size,
                              hipStream_t stream) {
  (void)in_sizes; (void)n_in; (void)out_size; (void)ws_size;
  const float* x        = (const float*)d_in[0];
  const float* ln1_g    = (const float*)d_in[1];
  const float* ln1_b    = (const float*)d_in[2];
  const float* ln2_g    = (const float*)d_in[3];
  const float* ln2_b    = (const float*)d_in[4];
  const float* in_proj  = (const float*)d_in[5];   // [2*DI, DM]
  const float* conv_w   = (const float*)d_in[6];   // [DI, 4]
  const float* conv_b   = (const float*)d_in[7];   // [DI]
  const float* x_proj   = (const float*)d_in[8];   // [96, DI]
  const float* dt_proj  = (const float*)d_in[9];   // [DI, 64]
  const float* dt_b     = (const float*)d_in[10];  // [DI]
  const float* A_log    = (const float*)d_in[11];  // [DI, 16]
  const float* Dp       = (const float*)d_in[12];  // [DI]
  const float* out_proj = (const float*)d_in[13];  // [DM, DI]
  float* out = (float*)d_out;

  // workspace layout
  char* p = (char*)d_ws;
  auto take = [&](size_t bytes) { char* r = p; p += (bytes + 255) & ~(size_t)255; return r; };
  float*  xz   = (float*) take((size_t)NTOK * 2 * DI * 4);
  float*  xcf  = (float*) take((size_t)NTOK * DI * 4);
  float*  dblf = (float*) take((size_t)NTOK * XPN * 4);
  float*  dt   = (float*) take((size_t)NTOK * DI * 4);
  __bf16* ub   = (__bf16*)take((size_t)NTOK * DM * 2);
  __bf16* xcb  = (__bf16*)take((size_t)NTOK * DI * 2);
  __bf16* dblb = (__bf16*)take((size_t)NTOK * XPN * 2);
  __bf16* yb   = (__bf16*)take((size_t)NTOK * DI * 2);
  __bf16* wInb = (__bf16*)take((size_t)2 * DI * DM * 2);
  __bf16* wXpb = (__bf16*)take((size_t)XPN * DI * 2);
  __bf16* wDtb = (__bf16*)take((size_t)DI * DTR * 2);
  __bf16* wOutb= (__bf16*)take((size_t)DM * DI * 2);

  auto cvt = [&](const float* s, __bf16* d, size_t n) {
    cvt_bf16_kernel<<<(unsigned)((n / 4 + 255) / 256 + 1), 256, 0, stream>>>(s, d, n);
  };
  // weights -> bf16 (once per call; deterministic)
  cvt(in_proj,  wInb,  (size_t)2 * DI * DM);
  cvt(x_proj,   wXpb,  (size_t)XPN * DI);
  cvt(dt_proj,  wDtb,  (size_t)DI * DTR);
  cvt(out_proj, wOutb, (size_t)DM * DI);

  const size_t nxe = (size_t)NTOK * DM;
  copy_kernel<<<(unsigned)((nxe + 255) / 256), 256, 0, stream>>>(x, out, nxe);

  for (int dir = 0; dir < 2; ++dir) {
    const float* g = dir ? ln2_g : ln1_g;
    const float* b = dir ? ln2_b : ln1_b;

    layernorm_kernel<<<NTOK, 256, 0, stream>>>(x, g, b, ub);

    // xz = u @ in_proj^T : M=8192, N=4096, K=1024
    gemm_wmma_bf16<EP_STORE, false><<<dim3((2 * DI) / BN, NTOK / BM), 256, 0, stream>>>(
        ub, wInb, xz, nullptr, nullptr, NTOK, 2 * DI, DM, DM);

    conv_silu_kernel<<<(unsigned)(((size_t)NTOK * DI) / 256), 256, 0, stream>>>(
        xz, conv_w, conv_b, xcf, xcb, dir);

    // dbl = xc @ x_proj^T : M=8192, N=96, K=2048 (+ bf16 mirror for dt GEMM)
    gemm_wmma_bf16<EP_STORE, true><<<dim3((XPN + BN - 1) / BN, NTOK / BM), 256, 0, stream>>>(
        xcb, wXpb, dblf, dblb, nullptr, NTOK, XPN, DI, DI);

    // dt = softplus(dbl[:, :64] @ dt_proj^T + dt_b) : M=8192, N=2048, K=64, lda=96
    gemm_wmma_bf16<EP_SOFTPLUS, false><<<dim3(DI / BN, NTOK / BM), 256, 0, stream>>>(
        dblb, wDtb, dt, nullptr, dt_b, NTOK, DI, DTR, XPN);

    scan_kernel<<<BATCH * (DI / 256), 256, 0, stream>>>(
        dt, dblf, xcf, xz, A_log, Dp, yb, dir);

    // out += y @ out_proj^T : M=8192, N=1024, K=2048
    gemm_wmma_bf16<EP_ACCUM, false><<<dim3(DM / BN, NTOK / BM), 256, 0, stream>>>(
        yb, wOutb, out, nullptr, nullptr, NTOK, DM, DI, DI);
  }
}
